// AdjacencyLearning_88261577932938
// MI455X (gfx1250) — matrix operand order repaired and checked by reading, not compiled, not run on hardware
//
#include <hip/hip_runtime.h>
#include <hip/hip_bf16.h>

typedef __attribute__((ext_vector_type(16))) _Float16 v16h;
typedef __attribute__((ext_vector_type(8)))  float    v8f;

#define N_NODES  50000
#define N_FEAT   128
#define N_EDGES  600000
#define N_POWERS 3

__device__ __forceinline__ float lane_bcast(float v, int srcLane) {
    // wave32 lane gather via LDS-permute hardware (no LDS memory touched)
    return __int_as_float(__builtin_amdgcn_ds_bpermute(srcLane << 2, __float_as_int(v)));
}

// |a-b| without the fabs bitmask idiom (works around gfx1250 ISel crash on
// v2f32-vectorized fabs): max(a-b, b-a) == |a-b| for finite inputs, lowers
// to v_sub_f32 + v_max_num_f32.
__device__ __forceinline__ float absdiff1(float a, float b) {
    return fmaxf(a - b, b - a);
}

__device__ __forceinline__ float4 absdiff4(float4 a, float4 b) {
    float4 r;
    r.x = absdiff1(a.x, b.x);
    r.y = absdiff1(a.y, b.y);
    r.z = absdiff1(a.z, b.z);
    r.w = absdiff1(a.w, b.w);
    return r;
}

__global__ __launch_bounds__(256)
void AdjacencyLearning_wmma_kernel(const float* __restrict__ x,
                                   const long long* __restrict__ edge_index,
                                   const float* __restrict__ Wa,
                                   const float* __restrict__ ba,
                                   const float* __restrict__ Wb,
                                   const float* __restrict__ bb,
                                   float* __restrict__ out)
{
    const int p    = blockIdx.y;            // power index 0..2
    const int lane = threadIdx.x & 31;
    const int N    = lane & 15;             // column (hidden unit) 0..15, valid 0..2
    const int hi   = (lane >> 4) & 1;       // half-wave select

    // ---- Build B = Wa[p] in WMMA 16-bit B layout, 4 K-chunks of 32 (loaded once) ----
    // B (32x16): lane half h holds K=8h..8h+7 (VGPR 0-3) and K=16+8h..16+8h+7 (VGPR 4-7)
    const float* WaP = Wa + p * (N_FEAT * 3);
    v16h B[4];
    #pragma unroll
    for (int c = 0; c < 4; ++c) {
        #pragma unroll
        for (int v = 0; v < 8; ++v) {
            const int f = c * 32 + (v >> 2) * 16 + hi * 8 + 2 * (v & 3);     // feature (K) index
            const float b0 = (N < 3) ? WaP[(f + 0) * 3 + N] : 0.0f;
            const float b1 = (N < 3) ? WaP[(f + 1) * 3 + N] : 0.0f;
            B[c][2 * v + 0] = (_Float16)b0;
            B[c][2 * v + 1] = (_Float16)b1;
        }
    }
    const float ban = (N < 3) ? ba[p * 3 + N] : 0.0f;   // hidden bias for this lane's column
    const float wbn = (N < 3) ? Wb[p * 3 + N] : 0.0f;   // output weight for this lane's column
    const float bbp = bb[p];

    const long long* ej = edge_index + (size_t)p * 2 * N_EDGES;   // source nodes
    const long long* ei = ej + N_EDGES;                           // target nodes
    float* outP = out + (size_t)p * N_EDGES;

    const int TILES = N_EDGES / 16;                      // 37500, exact
    const int wid   = blockIdx.x * (blockDim.x >> 5) + (threadIdx.x >> 5);
    const int nwave = gridDim.x * (blockDim.x >> 5);

    for (int t = wid; t < TILES; t += nwave) {
        const int tb = t * 16;
        const int e  = tb + N;                           // this lane's edge (row M = N)
        const int j  = (int)ej[e];
        const int i  = (int)ei[e];
        const float4* xj = (const float4*)(x + (size_t)j * N_FEAT);
        const float4* xi = (const float4*)(x + (size_t)i * N_FEAT);

        // prefetch next tile's gather indices (emits global_prefetch_b8)
        if (t + nwave < TILES) {
            __builtin_prefetch(ej + e + (size_t)nwave * 16, 0, 1);
            __builtin_prefetch(ei + e + (size_t)nwave * 16, 0, 1);
        }

        v8f C = {};
        #pragma unroll
        for (int c = 0; c < 4; ++c) {
            // A (16x32 f16): lane row M=N; this lane needs chunk-relative
            // K = hi*8 .. hi*8+7 (VGPR 0-3) and K = 16+hi*8 .. 16+hi*8+7 (VGPR 4-7)
            const int f4 = (c * 32 + hi * 8) >> 2;       // float4 index of first group
            const float4 d0 = absdiff4(xj[f4 + 0], xi[f4 + 0]);
            const float4 d1 = absdiff4(xj[f4 + 1], xi[f4 + 1]);
            const float4 d2 = absdiff4(xj[f4 + 4], xi[f4 + 4]);
            const float4 d3 = absdiff4(xj[f4 + 5], xi[f4 + 5]);

            v16h A;
            A[0]  = (_Float16)d0.x;  A[1]  = (_Float16)d0.y;
            A[2]  = (_Float16)d0.z;  A[3]  = (_Float16)d0.w;
            A[4]  = (_Float16)d1.x;  A[5]  = (_Float16)d1.y;
            A[6]  = (_Float16)d1.z;  A[7]  = (_Float16)d1.w;
            A[8]  = (_Float16)d2.x;  A[9]  = (_Float16)d2.y;
            A[10] = (_Float16)d2.z;  A[11] = (_Float16)d2.w;
            A[12] = (_Float16)d3.x;  A[13] = (_Float16)d3.y;
            A[14] = (_Float16)d3.z;  A[15] = (_Float16)d3.w;

            // D(16x16,f32) += A(16x32,f16) x B(32x16,f16)
            C = __builtin_amdgcn_wmma_f32_16x16x32_f16(
                    false, A, false, B[c], (short)0, C, false, false);
        }

        // ---- epilogue: bias + ReLU + output projection + sigmoid ----
        // Lane holds D[m][N] for m = r + 8*hi (r = component 0..7).
        float res[8];
        #pragma unroll
        for (int r = 0; r < 8; ++r) {
            const float z = fmaxf(C[r] + ban, 0.0f) * wbn;   // relu(h)*Wb[N]
            // reduce over hidden units N = 0..2 within this half-wave
            const int base = hi << 4;
            const float s = lane_bcast(z, base + 0)
                          + lane_bcast(z, base + 1)
                          + lane_bcast(z, base + 2);
            const float zz = s + bbp;
            res[r] = 1.0f / (1.0f + __expf(-zz));
        }

        // lanes 0 and 16 hold full results for rows m=0..7 / m=8..15 -> 2x b128 stores
        if (N == 0) {
            float4* o = (float4*)(outP + tb + hi * 8);
            o[0] = make_float4(res[0], res[1], res[2], res[3]);
            o[1] = make_float4(res[4], res[5], res[6], res[7]);
        }
    }
}

extern "C" void kernel_launch(void* const* d_in, const int* in_sizes, int n_in,
                              void* d_out, int out_size, void* d_ws, size_t ws_size,
                              hipStream_t stream) {
    const float*     x          = (const float*)d_in[0];
    const long long* edge_index = (const long long*)d_in[1];   // int64 per reference
    const float*     Wa         = (const float*)d_in[2];
    const float*     ba         = (const float*)d_in[3];
    const float*     Wb         = (const float*)d_in[4];
    const float*     bb         = (const float*)d_in[5];
    float*           out        = (float*)d_out;

    // 640 blocks x 8 waves = 5120 waves per power; 37500 tiles -> ~7 tiles/wave,
    // enough to amortize the per-wave Wa-tile load and saturate the WGPs.
    dim3 grid(640, N_POWERS, 1);
    dim3 block(256, 1, 1);
    AdjacencyLearning_wmma_kernel<<<grid, block, 0, stream>>>(
        x, edge_index, Wa, ba, Wb, bb, out);
}
